// minGRU_8950711845679
// MI455X (gfx1250) — matrix operand order repaired and checked by reading, not compiled, not run on hardware
//
#include <hip/hip_runtime.h>
#include <hip/hip_bf16.h>

// ---------- problem sizes (from setup_inputs) ----------
#define BB   4
#define TT   8192
#define DD   512
#define MM   (BB*TT)      // 32768
#define KDIM DD           // 512
#define NN2  (2*DD)       // 1024

// ---------- GEMM tiling ----------
#define BLK_M 256         // 4 waves x 4 tiles of 16 rows
#define BLK_D 32          // hidden columns per block (gate cols d+512 handled too)
#define BK    32

// ---------- scan chunking ----------
#define NC 32
#define LCH (TT/NC)       // 256

typedef __attribute__((ext_vector_type(16))) __bf16       bf16x16;
typedef __attribute__((ext_vector_type(8)))  float        v8f;
typedef __attribute__((ext_vector_type(4)))  float        f32x4;
typedef __attribute__((ext_vector_type(4)))  unsigned int u32x4;
typedef __attribute__((ext_vector_type(2)))  unsigned int u32x2;

union Frag { u32x4 u2[2]; bf16x16 h; };

__device__ __forceinline__ unsigned short f2bf(float f) {
    unsigned int u = __float_as_uint(f);
    unsigned int r = u + 0x7FFFu + ((u >> 16) & 1u);   // round-to-nearest-even
    return (unsigned short)(r >> 16);
}
__device__ __forceinline__ unsigned int pk2bf(float a, float b) {
#if __has_builtin(__builtin_amdgcn_cvt_pk_bf16_f32)
    auto p = __builtin_amdgcn_cvt_pk_bf16_f32(a, b);   // single v_cvt_pk_bf16_f32
    return __builtin_bit_cast(unsigned int, p);
#else
    return (unsigned int)f2bf(a) | ((unsigned int)f2bf(b) << 16);
#endif
}

// =====================================================================
// P0a: x (fp32, [M,K]) -> xbf (bf16, [M,K]) ; 8 elements / thread
// =====================================================================
__global__ __launch_bounds__(256)
void cvt_x_bf16(const float* __restrict__ x, unsigned short* __restrict__ xbf) {
    size_t i8 = (size_t)(blockIdx.x * 256 + threadIdx.x) * 8;
    f32x4 a = *(const f32x4*)(x + i8);
    f32x4 b = *(const f32x4*)(x + i8 + 4);
    u32x4 pk = { pk2bf(a.x, a.y), pk2bf(a.z, a.w), pk2bf(b.x, b.y), pk2bf(b.z, b.w) };
    *(u32x4*)(xbf + i8) = pk;
}

// =====================================================================
// P0b: W (fp32, [K,N2]) -> Wt (bf16, K-major [N2,K])  (pre-transposed)
// =====================================================================
__global__ __launch_bounds__(256)
void cvt_w_bf16t(const float* __restrict__ W, unsigned short* __restrict__ Wt) {
    int idx = blockIdx.x * 256 + threadIdx.x;     // N2 * K/4
    int k4  = idx % (KDIM / 4);
    int n   = idx / (KDIM / 4);
    float w0 = W[(size_t)(4 * k4 + 0) * NN2 + n];
    float w1 = W[(size_t)(4 * k4 + 1) * NN2 + n];
    float w2 = W[(size_t)(4 * k4 + 2) * NN2 + n];
    float w3 = W[(size_t)(4 * k4 + 3) * NN2 + n];
    u32x2 pk = { pk2bf(w0, w1), pk2bf(w2, w3) };
    *(u32x2*)(Wt + (size_t)n * KDIM + 4 * k4) = pk;
}

// =====================================================================
// K1: hg = x @ W via v_wmma_f32_16x16x32_bf16. LDS-free: fragments loaded
//     straight from L2-resident bf16 copies. Each wave: 4 M-tiles x 4 N-tiles
//     (N-tiles j=0,1 hidden cols d, j=2,3 their gate cols d+512 -> epilogue
//     computes c = sigmoid(-gate), v = sigmoid(gate)*g(hidden) in-register).
// =====================================================================
__global__ __launch_bounds__(128)
void mingru_gemm_wmma(const unsigned short* __restrict__ xbf,
                      const unsigned short* __restrict__ Wt,
                      float* __restrict__ cbuf, float* __restrict__ vout) {
    const int tid   = threadIdx.x;          // 0..127 (4 waves)
    const int lane  = tid & 31;
    const int w     = tid >> 5;
    const int row16 = lane & 15;
    const int hi    = lane >> 4;

    const int mBase = blockIdx.x * BLK_M;   // 128 blocks in M
    const int n0    = blockIdx.y * BLK_D;   // 16 blocks in D

    v8f acc[4][4];
    #pragma unroll
    for (int i = 0; i < 4; ++i)
        #pragma unroll
        for (int j = 0; j < 4; ++j) acc[i][j] = (v8f){};

    // per-lane fragment base pointers (CDNA5 16-bit A/B VGPR striping)
    const unsigned short* aPtr[4];
    #pragma unroll
    for (int i = 0; i < 4; ++i)
        aPtr[i] = xbf + (size_t)(mBase + 64 * w + 16 * i + row16) * KDIM + 8 * hi;

    const unsigned short* bPtr[4];
    #pragma unroll
    for (int j = 0; j < 4; ++j) {
        int col = ((j < 2) ? (n0 + 16 * j) : (DD + n0 + 16 * (j - 2))) + row16;
        bPtr[j] = Wt + (size_t)col * KDIM + 8 * hi;
    }

    for (int k0 = 0; k0 < KDIM; k0 += BK) {
        Frag bf_[4];
        #pragma unroll
        for (int j = 0; j < 4; ++j) {
            bf_[j].u2[0] = *(const u32x4*)(bPtr[j] + k0);      // K = k0+8*hi ..+7
            bf_[j].u2[1] = *(const u32x4*)(bPtr[j] + k0 + 16);
        }
        #pragma unroll
        for (int i = 0; i < 4; ++i) {
            Frag af;
            af.u2[0] = *(const u32x4*)(aPtr[i] + k0);          // K = k0+8*hi ..+7
            af.u2[1] = *(const u32x4*)(aPtr[i] + k0 + 16);     // K = k0+16+8*hi ..+7
            #pragma unroll
            for (int j = 0; j < 4; ++j)
                acc[i][j] = __builtin_amdgcn_wmma_f32_16x16x32_bf16(
                    false, af.h, false, bf_[j].h, (short)0, acc[i][j], false, false);
        }
    }

    // ---- epilogue: hidden (j) pairs with gate (j+2) at identical lane/VGPR ----
    #pragma unroll
    for (int i = 0; i < 4; ++i) {
        #pragma unroll
        for (int j = 0; j < 2; ++j) {
            const int d = n0 + 16 * j + row16;
            #pragma unroll
            for (int r = 0; r < 8; ++r) {
                const int m = mBase + 64 * w + 16 * i + 8 * hi + r;
                float h = acc[i][j][r];
                float g = acc[i][j + 2][r];
                float c  = __builtin_amdgcn_rcpf(1.0f + __expf(g));    // sigmoid(-g)
                float z  = 1.0f - c;                                   // sigmoid(g)
                float sh = __builtin_amdgcn_rcpf(1.0f + __expf(-h));   // sigmoid(h)
                float gf = (h >= 0.0f) ? (h + 0.5f) : sh;              // v_cndmask
                cbuf[(size_t)m * DD + d] = c;
                vout[(size_t)m * DD + d] = z * gf;
            }
        }
    }
}

// =====================================================================
// K2: per-(b,4d,chunk) local reduce: A = prod c, S = local scan end value
// =====================================================================
__global__ __launch_bounds__(256)
void mingru_chunk_reduce(const float* __restrict__ c, const float* __restrict__ v,
                         float* __restrict__ Abuf, float* __restrict__ Sbuf) {
    int idx   = blockIdx.x * blockDim.x + threadIdx.x;   // B*NC*(D/4)
    int d4    = idx % (DD / 4);
    int bc    = idx / (DD / 4);                          // b*NC + chunk
    int chunk = bc % NC;
    int b     = bc / NC;
    size_t base = ((size_t)(b * TT + chunk * LCH)) * DD + 4 * d4;
    f32x4 a = {1.0f, 1.0f, 1.0f, 1.0f};
    f32x4 s = {};
    for (int t = 0; t < LCH; ++t) {
        f32x4 ct = *(const f32x4*)(c + base + (size_t)t * DD);
        f32x4 vt = *(const f32x4*)(v + base + (size_t)t * DD);
        s = ct * s + vt;
        a *= ct;
    }
    *(f32x4*)(Abuf + (size_t)bc * DD + 4 * d4) = a;
    *(f32x4*)(Sbuf + (size_t)bc * DD + 4 * d4) = s;
}

// =====================================================================
// K3: sequential carry across the 32 chunk aggregates per channel
// =====================================================================
__global__ __launch_bounds__(256)
void mingru_carry(const float* __restrict__ Abuf, const float* __restrict__ Sbuf,
                  float* __restrict__ Cin) {
    int idx = blockIdx.x * blockDim.x + threadIdx.x;     // B*(D/4)
    int d4 = idx % (DD / 4);
    int b  = idx / (DD / 4);
    f32x4 h = {};
    for (int j = 0; j < NC; ++j) {
        size_t o = ((size_t)(b * NC + j)) * DD + 4 * d4;
        *(f32x4*)(Cin + o) = h;
        h = (*(const f32x4*)(Abuf + o)) * h + (*(const f32x4*)(Sbuf + o));
    }
}

// =====================================================================
// K4: apply carries; recurrence h = c*h + v, overwrite v (== d_out) in place
// =====================================================================
__global__ __launch_bounds__(256)
void mingru_apply(const float* __restrict__ c, float* __restrict__ vout,
                  const float* __restrict__ Cin) {
    int idx   = blockIdx.x * blockDim.x + threadIdx.x;
    int d4    = idx % (DD / 4);
    int bc    = idx / (DD / 4);
    int chunk = bc % NC;
    int b     = bc / NC;
    size_t base = ((size_t)(b * TT + chunk * LCH)) * DD + 4 * d4;
    f32x4 h = *(const f32x4*)(Cin + (size_t)bc * DD + 4 * d4);
    for (int t = 0; t < LCH; ++t) {
        size_t o = base + (size_t)t * DD;
        f32x4 ct = *(const f32x4*)(c + o);
        f32x4 vt = *(f32x4*)(vout + o);
        h = ct * h + vt;
        *(f32x4*)(vout + o) = h;
    }
}

extern "C" void kernel_launch(void* const* d_in, const int* in_sizes, int n_in,
                              void* d_out, int out_size, void* d_ws, size_t ws_size,
                              hipStream_t stream) {
    const float* x  = (const float*)d_in[0];   // [4,8192,512] fp32
    const float* Wp = (const float*)d_in[1];   // [512,1024]   fp32
    float* out = (float*)d_out;                // [4,8192,512] fp32 (also holds v)

    float* wsf  = (float*)d_ws;
    float* cbuf = wsf;                                   // M*D floats (64 MB)
    float* Abuf = cbuf + (size_t)MM * DD;                // B*NC*D
    float* Sbuf = Abuf + (size_t)BB * NC * DD;
    float* Cin  = Sbuf + (size_t)BB * NC * DD;
    unsigned short* xbf = (unsigned short*)(Cin + (size_t)BB * NC * DD);  // M*K bf16 (32 MB)
    unsigned short* Wtbf = xbf + (size_t)MM * KDIM;                       // N2*K bf16 (1 MB)

    // pre-convert to bf16 (once; L2-resident afterwards)
    cvt_x_bf16<<<(MM * KDIM / 8) / 256, 256, 0, stream>>>(x, xbf);
    cvt_w_bf16t<<<(NN2 * (KDIM / 4)) / 256, 256, 0, stream>>>(Wp, Wtbf);

    dim3 g1(MM / BLK_M, DD / BLK_D);                     // (128,16)
    mingru_gemm_wmma<<<g1, 128, 0, stream>>>(xbf, Wtbf, cbuf, out);

    int nwork4 = BB * NC * (DD / 4);                     // 16384
    mingru_chunk_reduce<<<nwork4 / 256, 256, 0, stream>>>(cbuf, out, Abuf, Sbuf);
    mingru_carry<<<(BB * (DD / 4)) / 256, 256, 0, stream>>>(Abuf, Sbuf, Cin);
    mingru_apply<<<nwork4 / 256, 256, 0, stream>>>(cbuf, out, Cin);
}
